// Experts_20160576487899
// MI455X (gfx1250) — matrix operand order repaired and checked by reading, not compiled, not run on hardware
//
#include <hip/hip_runtime.h>
#include <cstdint>
#include <cstddef>

typedef __bf16 bf16;
typedef __attribute__((ext_vector_type(4)))  bf16  v4bf;
typedef __attribute__((ext_vector_type(8)))  bf16  v8bf;
typedef __attribute__((ext_vector_type(16))) bf16  v16bf;
typedef __attribute__((ext_vector_type(8)))  float v8f;

#define T_DIM 4096
#define H_DIM 1024
#define I_DIM 1024
#define E_DIM 8
#define F_DIM (2 * I_DIM)   // 2048
#define ALPHA_C 1.702f
#define LIMIT_C 7.0f

#define WMMA_BF16(a, b, c) \
  __builtin_amdgcn_wmma_f32_16x16x32_bf16(false, (a), false, (b), (short)0, (c), false, false)

// ---------------------------------------------------------------------------
// Fragment helpers (wave32, CDNA5 WMMA VGPR layouts, CDNA5 ISA).
// byteoff is a per-lane 32-bit BYTE offset (row*ld + klo already folded in),
// so loads lower to the saddr + 32-bit-voffset global_load_b128 form with the
// lo/hi split carried in the instruction's immediate offset.
// ---------------------------------------------------------------------------
__device__ __forceinline__ v16bf frag_join(v8bf lo, v8bf hi) {
  union { v16bf v; v8bf h[2]; } u;
  u.h[0] = lo; u.h[1] = hi;
  return u.v;
}

// A-matrix 16x32 bf16 (MxK) row-major: lane holds row M = lane&15,
// elems 0..7 = K[klo..klo+7], elems 8..15 = K[16+klo..23+klo], klo = lane<16?0:8.
// byteoff points at (row, k0 + klo); hi part is +16 elements = +32 bytes.
__device__ __forceinline__ v16bf load_fragA(const bf16* base, uint32_t byteoff) {
  const char* p = (const char*)base + byteoff;
  v8bf lo = *(const v8bf*)(p);
  v8bf hi = *(const v8bf*)(p + 32);
  return frag_join(lo, hi);
}

// B-matrix 32x16 bf16 (KxN), stored N-major (Bt[n][k]): lane holds col N=lane&15,
// elems 0..7 = K[klo..klo+7], elems 8..15 = K[16+klo..23+klo], klo = lane<16?0:8. byteoff points at
// (row, k0 + klo); hi part is +16 elements = +32 bytes.
__device__ __forceinline__ v16bf load_fragB(const bf16* base, uint32_t byteoff) {
  const char* p = (const char*)base + byteoff;
  v8bf lo = *(const v8bf*)(p);
  v8bf hi = *(const v8bf*)(p + 32);
  return frag_join(lo, hi);
}

// Exchange with neighbor lane (xor 1): ds_swizzle SWAPX1 (xor=1,or=0,and=0x1f)
__device__ __forceinline__ float swap_neighbor(float x) {
  int i = __builtin_bit_cast(int, x);
  i = __builtin_amdgcn_ds_swizzle(i, 0x041F);
  return __builtin_bit_cast(float, i);
}

// ---------------------------------------------------------------------------
// Prologue: fp32 -> bf16 elementwise (n must be a multiple of 4)
// ---------------------------------------------------------------------------
__global__ __launch_bounds__(256)
void cvt_f32_bf16(const float* __restrict__ in, bf16* __restrict__ out, int n) {
  int i = (blockIdx.x * 256 + threadIdx.x) * 4;
  if (i + 3 < n) {
    float4 v = *(const float4*)(in + i);
    v4bf o;
    o[0] = (bf16)v.x; o[1] = (bf16)v.y; o[2] = (bf16)v.z; o[3] = (bf16)v.w;
    *(v4bf*)(out + i) = o;
  }
}

// ---------------------------------------------------------------------------
// Prologue: gate_up_proj (E,H,F) fp32 -> (E,F,H) bf16, tiled transpose via LDS
// ---------------------------------------------------------------------------
__global__ __launch_bounds__(256)
void transpose_cvt(const float* __restrict__ in, bf16* __restrict__ out) {
  __shared__ float tile[32][33];
  int e  = blockIdx.z;
  int f0 = blockIdx.x * 32;
  int h0 = blockIdx.y * 32;
  const float* src = in  + (size_t)e * H_DIM * F_DIM;
  bf16*        dst = out + (size_t)e * F_DIM * H_DIM;
  int tx = threadIdx.x & 31;
  int ty = threadIdx.x >> 5;     // 0..7
  #pragma unroll
  for (int j = ty; j < 32; j += 8)
    tile[j][tx] = src[(size_t)(h0 + j) * F_DIM + (f0 + tx)];
  __syncthreads();
  #pragma unroll
  for (int j = ty; j < 32; j += 8)
    dst[(size_t)(f0 + j) * H_DIM + (h0 + tx)] = (bf16)tile[tx][j];
}

// ---------------------------------------------------------------------------
// GEMM1 + bias + gated activation: act[e,t,i], bf16 out.
// Wave tile: 32(t) x 64(f). K-loop is a 2-stage ping-pong (no register
// rotation copies). Block = 4 waves (2x2) -> 64t x 128f.
// Grid: (T/64, F/128, E) = (64, 16, 8)
// ---------------------------------------------------------------------------
__global__ __launch_bounds__(128)
void gemm_gate_up(const bf16*  __restrict__ hid,   // (T,H) bf16
                  const bf16*  __restrict__ wt,    // (E,F,H) bf16 (pre-transposed)
                  const float* __restrict__ gub,   // (E,F) fp32 bias
                  bf16*        __restrict__ act)   // (E,T,I) bf16
{
  const int lane  = threadIdx.x & 31;
  const int wave  = threadIdx.x >> 5;
  const int wm    = wave & 1;          // wave M-group (t)
  const int wn    = wave >> 1;         // wave N-group (f)
  const int e     = blockIdx.z;
  const int tbase = blockIdx.x * 64 + wm * 32;
  const int fbase = blockIdx.y * 128 + wn * 64;
  const int m     = lane & 15;
  const int hi    = (lane >> 4) & 1;
  const uint32_t kloA = (lane & 16) ? 8u : 0u;
  const uint32_t kloB = (lane & 16) ? 8u : 0u;

  const bf16* B = wt + (size_t)e * F_DIM * H_DIM;   // uniform base

  // Per-lane byte offsets (klo folded in once); advance by 128 B per 2 K-steps.
  uint32_t aoff0 = ((uint32_t)(tbase + m) * H_DIM + kloA) * 2u;
  uint32_t aoff1 = aoff0 + 16u * H_DIM * 2u;
  uint32_t boff0 = ((uint32_t)(fbase + 0 * 16 + m) * H_DIM + kloB) * 2u;
  uint32_t boff1 = ((uint32_t)(fbase + 1 * 16 + m) * H_DIM + kloB) * 2u;
  uint32_t boff2 = ((uint32_t)(fbase + 2 * 16 + m) * H_DIM + kloB) * 2u;
  uint32_t boff3 = ((uint32_t)(fbase + 3 * 16 + m) * H_DIM + kloB) * 2u;

  v8f acc[2][4] = {};

  v16bf a0 = load_fragA(hid, aoff0);
  v16bf a1 = load_fragA(hid, aoff1);
  v16bf b0 = load_fragB(B, boff0);
  v16bf b1 = load_fragB(B, boff1);
  v16bf b2 = load_fragB(B, boff2);
  v16bf b3 = load_fragB(B, boff3);

  #pragma unroll 1
  for (uint32_t k = 0; k < H_DIM; k += 64) {
    // ---- stage 0: consume set X (k), prefetch set Y (k+32 -> byte +64)
    v16bf a0n = load_fragA(hid, aoff0 + 64u);
    v16bf a1n = load_fragA(hid, aoff1 + 64u);
    v16bf b0n = load_fragB(B, boff0 + 64u);
    v16bf b1n = load_fragB(B, boff1 + 64u);
    v16bf b2n = load_fragB(B, boff2 + 64u);
    v16bf b3n = load_fragB(B, boff3 + 64u);

    acc[0][0] = WMMA_BF16(a0, b0, acc[0][0]);
    acc[1][0] = WMMA_BF16(a1, b0, acc[1][0]);
    acc[0][1] = WMMA_BF16(a0, b1, acc[0][1]);
    acc[1][1] = WMMA_BF16(a1, b1, acc[1][1]);
    acc[0][2] = WMMA_BF16(a0, b2, acc[0][2]);
    acc[1][2] = WMMA_BF16(a1, b2, acc[1][2]);
    acc[0][3] = WMMA_BF16(a0, b3, acc[0][3]);
    acc[1][3] = WMMA_BF16(a1, b3, acc[1][3]);

    // ---- stage 1: consume set Y (k+32), refill set X (k+64 -> byte +128).
    // Final iteration over-reads <=208 B past the logical buffer end; the
    // workspace layout pads every buffer with 4 KiB, so this stays in-bounds.
    a0 = load_fragA(hid, aoff0 + 128u);
    a1 = load_fragA(hid, aoff1 + 128u);
    b0 = load_fragB(B, boff0 + 128u);
    b1 = load_fragB(B, boff1 + 128u);
    b2 = load_fragB(B, boff2 + 128u);
    b3 = load_fragB(B, boff3 + 128u);

    acc[0][0] = WMMA_BF16(a0n, b0n, acc[0][0]);
    acc[1][0] = WMMA_BF16(a1n, b0n, acc[1][0]);
    acc[0][1] = WMMA_BF16(a0n, b1n, acc[0][1]);
    acc[1][1] = WMMA_BF16(a1n, b1n, acc[1][1]);
    acc[0][2] = WMMA_BF16(a0n, b2n, acc[0][2]);
    acc[1][2] = WMMA_BF16(a1n, b2n, acc[1][2]);
    acc[0][3] = WMMA_BF16(a0n, b3n, acc[0][3]);
    acc[1][3] = WMMA_BF16(a1n, b3n, acc[1][3]);

    aoff0 += 128u; aoff1 += 128u;
    boff0 += 128u; boff1 += 128u; boff2 += 128u; boff3 += 128u;
  }

  // Epilogue: C layout -> lane holds (M = r + 8*hi, N = m). Column parity == m
  // parity: even lane = gate (2i), odd neighbor = up (2i+1).
  #pragma unroll
  for (int mt = 0; mt < 2; ++mt) {
    #pragma unroll
    for (int nt = 0; nt < 4; ++nt) {
      int   f    = fbase + nt * 16 + m;
      float bias = gub[(size_t)e * F_DIM + f];
      #pragma unroll
      for (int r = 0; r < 8; ++r) {
        float c     = acc[mt][nt][r] + bias;
        float other = swap_neighbor(c);          // uniform: all lanes execute
        if ((m & 1) == 0) {
          float gate = fminf(c, LIMIT_C);
          float up   = fminf(fmaxf(other, -LIMIT_C), LIMIT_C);
          float glu  = gate / (1.0f + __expf(-ALPHA_C * gate));
          float a    = (up + 1.0f) * glu;
          int   t    = tbase + mt * 16 + r + 8 * hi;
          int   i    = f >> 1;
          act[((size_t)e * T_DIM + t) * I_DIM + i] = (bf16)a;
        }
      }
    }
  }
}

// ---------------------------------------------------------------------------
// GEMM2 + bias + routing-weighted expert sum: out[t,h] fp32.
// Wave tile: 16(t) x 64(h), 2-stage ping-pong K-loop; loops all E experts.
// Block = 4 waves along t -> 64t x 64h.  Grid: (T/64, H/64) = (64, 16)
// ---------------------------------------------------------------------------
__global__ __launch_bounds__(128)
void gemm_down_combine(const bf16*  __restrict__ act,   // (E,T,I) bf16
                       const bf16*  __restrict__ dw,    // (E,H,I) bf16
                       const float* __restrict__ dbias, // (E,H)
                       const float* __restrict__ rw,    // (T,E)
                       float*       __restrict__ out)   // (T,H)
{
  const int lane  = threadIdx.x & 31;
  const int wave  = threadIdx.x >> 5;
  const int tbase = blockIdx.x * 64 + wave * 16;
  const int hbase = blockIdx.y * 64;
  const int m     = lane & 15;
  const int hi    = (lane >> 4) & 1;
  const uint32_t kloA = (lane & 16) ? 8u : 0u;
  const uint32_t kloB = (lane & 16) ? 8u : 0u;

  const uint32_t aoff_base = ((uint32_t)(tbase + m) * I_DIM + kloA) * 2u;
  const uint32_t boff_base0 = ((uint32_t)(hbase + 0 * 16 + m) * I_DIM + kloB) * 2u;
  const uint32_t boff_base1 = ((uint32_t)(hbase + 1 * 16 + m) * I_DIM + kloB) * 2u;
  const uint32_t boff_base2 = ((uint32_t)(hbase + 2 * 16 + m) * I_DIM + kloB) * 2u;
  const uint32_t boff_base3 = ((uint32_t)(hbase + 3 * 16 + m) * I_DIM + kloB) * 2u;

  float outacc[4][8] = {};

  #pragma unroll 1
  for (int e = 0; e < E_DIM; ++e) {
    const bf16* A = act + (size_t)e * T_DIM * I_DIM;  // uniform base
    const bf16* B = dw  + (size_t)e * H_DIM * I_DIM;  // uniform base

    uint32_t aoff  = aoff_base;
    uint32_t boff0 = boff_base0, boff1 = boff_base1;
    uint32_t boff2 = boff_base2, boff3 = boff_base3;

    v8f acc[4] = {};
    v16bf a  = load_fragA(A, aoff);
    v16bf b0 = load_fragB(B, boff0);
    v16bf b1 = load_fragB(B, boff1);
    v16bf b2 = load_fragB(B, boff2);
    v16bf b3 = load_fragB(B, boff3);

    #pragma unroll 1
    for (uint32_t k = 0; k < I_DIM; k += 64) {
      // stage 0: consume (k), prefetch (k+32)
      v16bf an  = load_fragA(A, aoff + 64u);
      v16bf b0n = load_fragB(B, boff0 + 64u);
      v16bf b1n = load_fragB(B, boff1 + 64u);
      v16bf b2n = load_fragB(B, boff2 + 64u);
      v16bf b3n = load_fragB(B, boff3 + 64u);

      acc[0] = WMMA_BF16(a, b0, acc[0]);
      acc[1] = WMMA_BF16(a, b1, acc[1]);
      acc[2] = WMMA_BF16(a, b2, acc[2]);
      acc[3] = WMMA_BF16(a, b3, acc[3]);

      // stage 1: consume (k+32), refill (k+64) (tail over-read lands in pad)
      a  = load_fragA(A, aoff + 128u);
      b0 = load_fragB(B, boff0 + 128u);
      b1 = load_fragB(B, boff1 + 128u);
      b2 = load_fragB(B, boff2 + 128u);
      b3 = load_fragB(B, boff3 + 128u);

      acc[0] = WMMA_BF16(an, b0n, acc[0]);
      acc[1] = WMMA_BF16(an, b1n, acc[1]);
      acc[2] = WMMA_BF16(an, b2n, acc[2]);
      acc[3] = WMMA_BF16(an, b3n, acc[3]);

      aoff += 128u;
      boff0 += 128u; boff1 += 128u; boff2 += 128u; boff3 += 128u;
    }

    float w[8];
    #pragma unroll
    for (int r = 0; r < 8; ++r)
      w[r] = rw[(size_t)(tbase + r + 8 * hi) * E_DIM + e];

    #pragma unroll
    for (int nt = 0; nt < 4; ++nt) {
      float db = dbias[(size_t)e * H_DIM + hbase + nt * 16 + m];
      #pragma unroll
      for (int r = 0; r < 8; ++r)
        outacc[nt][r] += w[r] * (acc[nt][r] + db);
    }
  }

  #pragma unroll
  for (int nt = 0; nt < 4; ++nt)
    #pragma unroll
    for (int r = 0; r < 8; ++r)
      out[(size_t)(tbase + r + 8 * hi) * H_DIM + hbase + nt * 16 + m] = outacc[nt][r];
}

// ---------------------------------------------------------------------------
// Launch
// ---------------------------------------------------------------------------
extern "C" void kernel_launch(void* const* d_in, const int* in_sizes, int n_in,
                              void* d_out, int out_size, void* d_ws, size_t ws_size,
                              hipStream_t stream) {
  (void)in_sizes; (void)n_in; (void)out_size; (void)ws_size;

  const float* hidden  = (const float*)d_in[0];  // (T,H)
  const float* routing = (const float*)d_in[1];  // (T,E)
  const float* gup     = (const float*)d_in[2];  // (E,H,2I)
  const float* gup_b   = (const float*)d_in[3];  // (E,2I)
  const float* down    = (const float*)d_in[4];  // (E,H,I)
  const float* down_b  = (const float*)d_in[5];  // (E,I) == (E,H) here
  float*       out     = (float*)d_out;          // (T,H)

  // Workspace layout: each buffer followed by a 4 KiB pad for tail prefetch.
  const size_t PAD = 4096;
  char* ws = (char*)d_ws;
  size_t off = 0;
  bf16* hid_bf = (bf16*)(ws + off); off += (size_t)T_DIM * H_DIM * 2 + PAD;          //  8 MiB
  bf16* wt_bf  = (bf16*)(ws + off); off += (size_t)E_DIM * F_DIM * H_DIM * 2 + PAD;  // 32 MiB
  bf16* dw_bf  = (bf16*)(ws + off); off += (size_t)E_DIM * H_DIM * I_DIM * 2 + PAD;  // 16 MiB
  bf16* act_bf = (bf16*)(ws + off); off += (size_t)E_DIM * T_DIM * I_DIM * 2 + PAD;  // 64 MiB

  // 1) convert hidden_states: T*H = 4,194,304 elems -> 4096 blocks
  cvt_f32_bf16<<<dim3(4096), dim3(256), 0, stream>>>(hidden, hid_bf, T_DIM * H_DIM);

  // 2) convert down_proj: E*H*I = 8,388,608 elems -> 8192 blocks
  cvt_f32_bf16<<<dim3(8192), dim3(256), 0, stream>>>(down, dw_bf, E_DIM * H_DIM * I_DIM);

  // 3) transpose+convert gate_up_proj: (E,H,F) -> (E,F,H)
  transpose_cvt<<<dim3(F_DIM / 32, H_DIM / 32, E_DIM), dim3(256), 0, stream>>>(gup, wt_bf);

  // 4) GEMM1 + activation -> act (E,T,I) bf16
  gemm_gate_up<<<dim3(T_DIM / 64, F_DIM / 128, E_DIM), dim3(128), 0, stream>>>(
      hid_bf, wt_bf, gup_b, act_bf);

  // 5) GEMM2 + bias + routing combine -> out (T,H) fp32
  gemm_down_combine<<<dim3(T_DIM / 64, H_DIM / 64), dim3(128), 0, stream>>>(
      act_bf, dw_bf, down_b, routing, out);
}
